// EncoderDecoderAttention_19301583028270
// MI455X (gfx1250) — matrix operand order repaired and checked by reading, not compiled, not run on hardware
//
#include <hip/hip_runtime.h>

typedef _Float16 half_t;
typedef _Float16 v8h  __attribute__((ext_vector_type(8)));
typedef _Float16 v16h __attribute__((ext_vector_type(16)));
typedef float    v8f  __attribute__((ext_vector_type(8)));
typedef float    v4f  __attribute__((ext_vector_type(4)));
typedef unsigned int v4u __attribute__((ext_vector_type(4)));
typedef int      v8i  __attribute__((ext_vector_type(8)));
typedef int      v4i  __attribute__((ext_vector_type(4)));

#define EMBED    1024
#define HEADS    16
#define HEAD_DIM 64
#define BATCH    2
#define SQ       2048
#define SK       2048

union V16U { v16h v; v8h h[2]; };

static __device__ inline v16h ld16(const half_t* p0, const half_t* p1) {
    V16U u;
    u.h[0] = *(const v8h*)p0;
    u.h[1] = *(const v8h*)p1;
    return u.v;
}

static __device__ inline v8f zero8() {
    v8f z;
    #pragma unroll
    for (int i = 0; i < 8; ++i) z[i] = 0.0f;
    return z;
}

// Read 16 contiguous f32 (16B aligned), convert to f16, store as 2x ds_store_b128.
static __device__ inline void cvt_store16(const float* __restrict__ src, half_t* dst) {
    v4f a0 = *(const v4f*)(src);
    v4f a1 = *(const v4f*)(src + 4);
    v4f a2 = *(const v4f*)(src + 8);
    v4f a3 = *(const v4f*)(src + 12);
    v8h h0, h1;
    #pragma unroll
    for (int i = 0; i < 4; ++i) {
        h0[i]     = (half_t)a0[i];
        h0[4 + i] = (half_t)a1[i];
        h1[i]     = (half_t)a2[i];
        h1[4 + i] = (half_t)a3[i];
    }
    *(v8h*)dst       = h0;
    *(v8h*)(dst + 8) = h1;
}

// ---------------------------------------------------------------------------
// Tensor Data Mover: 2-D f16 tile DMA global -> LDS with row padding.
// D# layout per CDNA5 ISA §8.3/8.4. data_size=2B. pad codes: interval
// 0=2,1=4,2=8,3=16,4=32.. DWORDs; amount 0=1,1=2,...  (both always enabled).
// This toolchain exposes the 6-arg builtin:
//   (uint32x4 g0, int32x8 g1, int32x4 g2, int32x4 g3, int32x8, i32 cpol)
// ---------------------------------------------------------------------------
static __device__ inline void tdm_load_2d_f16(const half_t* gptr, unsigned lds_addr,
                                              unsigned dim0, unsigned dim1,
                                              unsigned tile0, unsigned tile1,
                                              unsigned long long stride0,
                                              unsigned pad_interval, unsigned pad_amount)
{
    unsigned long long ga = (unsigned long long)(size_t)gptr;
    v4u g0;
    g0[0] = 1u;                                         // count=1, user descriptor
    g0[1] = lds_addr;                                   // LDS byte address
    g0[2] = (unsigned)ga;                               // global_addr[31:0]
    g0[3] = (unsigned)(ga >> 32) | (2u << 30);          // global_addr[56:32] | type=2
    v8i g1;
    g1[0] = (int)((1u << 16)                            // data_size = 2 bytes
                | (1u << 20)                            // pad_enable
                | (pad_interval << 22) | (pad_amount << 25));
    g1[1] = (int)((dim0 & 0xFFFFu) << 16);              // tensor_dim0[15:0] @ bits 63:48
    g1[2] = (int)((dim0 >> 16) | ((dim1 & 0xFFFFu) << 16));
    g1[3] = (int)((dim1 >> 16) | (tile0 << 16));        // tile_dim0 @ bits 127:112
    g1[4] = (int)(tile1 & 0xFFFFu);                     // tile_dim1; tile_dim2=0
    g1[5] = (int)(unsigned)stride0;                     // tensor_dim0_stride[31:0]
    g1[6] = (int)(unsigned)(stride0 >> 32);             // stride0[47:32]; dim1_stride=0
    g1[7] = 0;
    v4i z4; z4[0] = 0; z4[1] = 0; z4[2] = 0; z4[3] = 0; // groups 2/3 unused (2-D)
    v8i z8;
    #pragma unroll
    for (int i = 0; i < 8; ++i) z8[i] = 0;
    __builtin_amdgcn_tensor_load_to_lds(g0, g1, z4, z4, z8, 0);
}

// ---------------------------------------------------------------------------
// Kernel 1: per-head projection  Out[b,h] = X[b] (SxE) * W[h] (Ex64) + bias[h]
// transpose_out==0 : Out layout [B*H][S][64]   (Q, K)
// transpose_out==1 : Out layout [B*H][64][S]   (V transposed, ready as B-frag)
// ---------------------------------------------------------------------------
__global__ __launch_bounds__(128)
void proj_kernel(const float* __restrict__ X, const float* __restrict__ W,
                 const float* __restrict__ bias, half_t* __restrict__ Out,
                 int S, int transpose_out)
{
    __shared__ half_t As[64 * 40];   // A tile [m][k], stride 40 halves (80B rows)
    __shared__ half_t Bt[64 * 40];   // B^T tile [n][k]

    const int tid = threadIdx.x;
    const int wave = tid >> 5, lane = tid & 31;
    const int half16 = lane >> 4, l16 = lane & 15;

    const int mtile = blockIdx.x;          // S/64 tiles
    const int bh = blockIdx.y;             // b*HEADS + h
    const int h = bh % HEADS;
    const int b = bh / HEADS;

    const float* Xb = X + (size_t)b * S * EMBED + (size_t)mtile * 64 * EMBED;
    const float* Wh = W + (size_t)h * EMBED * HEAD_DIM;

    v8f acc[4];
    #pragma unroll
    for (int i = 0; i < 4; ++i) acc[i] = zero8();

    for (int k0 = 0; k0 < EMBED; k0 += 32) {
        {   // stage A: 64x32 f32 -> f16, row-major; vectorized b128 loads/stores
            int r = tid >> 1, cc = (tid & 1) * 16;
            cvt_store16(Xb + (size_t)r * EMBED + k0 + cc, &As[r * 40 + cc]);
            if (k0 + 32 < EMBED)   // prefetch next A chunk into caches
                __builtin_prefetch(Xb + (size_t)r * EMBED + k0 + 32 + cc, 0, 1);
        }
        {   // stage B^T: thread owns column n; coalesced strided reads,
            // packed contiguous ds_store_b128 x2 writes
            int n = tid & 63, rh = tid >> 6;            // rh=0/1 -> k rows 0..15/16..31
            const float* src = Wh + (size_t)(k0 + rh * 16) * HEAD_DIM + n;
            v8h h0, h1;
            #pragma unroll
            for (int j = 0; j < 8; ++j) h0[j] = (half_t)src[(size_t)j * HEAD_DIM];
            #pragma unroll
            for (int j = 0; j < 8; ++j) h1[j] = (half_t)src[(size_t)(8 + j) * HEAD_DIM];
            *(v8h*)&Bt[n * 40 + rh * 16]     = h0;
            *(v8h*)&Bt[n * 40 + rh * 16 + 8] = h1;
            if (k0 + 32 < EMBED)
                __builtin_prefetch(src + 32 * HEAD_DIM, 0, 1);
        }
        __syncthreads();

        int mrow = wave * 16 + l16;
        // A frag: lanes<16: K{0..7,16..23}; lanes>=16: K{8..15,24..31}
        v16h af = ld16(&As[mrow * 40 + half16 * 8], &As[mrow * 40 + 16 + half16 * 8]);
        v16h bf[4];
        #pragma unroll
        for (int ns = 0; ns < 4; ++ns) {
            int ncol = ns * 16 + l16;
            bf[ns] = ld16(&Bt[ncol * 40 + half16 * 8], &Bt[ncol * 40 + 16 + half16 * 8]);
        }
        #pragma unroll
        for (int ns = 0; ns < 4; ++ns)
            acc[ns] = __builtin_amdgcn_wmma_f32_16x16x32_f16(
                false, af, false, bf[ns], (short)0, acc[ns], false, false);
        __syncthreads();
    }

    const float* bia = bias + h * HEAD_DIM;
    if (!transpose_out) {
        half_t* O = Out + ((size_t)bh * S + mtile * 64 + wave * 16) * HEAD_DIM;
        #pragma unroll
        for (int ns = 0; ns < 4; ++ns) {
            int n = ns * 16 + l16;
            float bb = bia[n];
            #pragma unroll
            for (int v = 0; v < 8; ++v) {
                int m = v + half16 * 8;
                O[m * HEAD_DIM + n] = (half_t)(acc[ns][v] + bb);
            }
        }
    } else {
        // Out[bh][d][S]; lane's 8 row values contiguous in m -> b128 store
        #pragma unroll
        for (int ns = 0; ns < 4; ++ns) {
            int n = ns * 16 + l16;
            float bb = bia[n];
            v8h pk;
            #pragma unroll
            for (int v = 0; v < 8; ++v) pk[v] = (half_t)(acc[ns][v] + bb);
            half_t* O = Out + ((size_t)bh * HEAD_DIM + n) * S
                            + mtile * 64 + wave * 16 + half16 * 8;
            *(v8h*)O = pk;
        }
    }
}

// ---------------------------------------------------------------------------
// Kernel 2: flash attention per (b,h,qtile). 4 waves x 16 q-rows.
// K/V^T chunks move via the Tensor Data Mover, double-buffered in LDS.
// ---------------------------------------------------------------------------
__global__ __launch_bounds__(128)
void attn_kernel(const half_t* __restrict__ Qb,   // [B*H][SQ][64]
                 const half_t* __restrict__ Kb,   // [B*H][SK][64]
                 const half_t* __restrict__ Vt,   // [B*H][64][SK]
                 half_t* __restrict__ Ctx)        // [B][SQ][EMBED] head-major
{
    __shared__ half_t Kbuf[2][32 * 72];    // K rows [k][d], stride 72 (pad 4DW/32DW)
    __shared__ half_t Vbuf[2][64 * 40];    // V^T rows [d][k], stride 40 (pad 4DW/16DW)
    __shared__ half_t Pbuf[4 * 16 * 40];   // per-wave P tile [m][k]

    const int tid = threadIdx.x;
    const int wave = tid >> 5, lane = tid & 31;
    const int half16 = lane >> 4, l16 = lane & 15;
    const int qtile = blockIdx.x, bh = blockIdx.y;
    const int h = bh % HEADS;
    const int b = bh / HEADS;

    const half_t* Qw = Qb + ((size_t)bh * SQ + qtile * 64 + wave * 16) * HEAD_DIM;
    const half_t* Kh = Kb + (size_t)bh * SK * HEAD_DIM;
    const half_t* Vh = Vt + (size_t)bh * HEAD_DIM * SK;

    // Q A-fragments (d-chunks 0..31, 32..63) kept in registers for whole loop
    v16h qf[2];
    {
        const half_t* qp = Qw + l16 * HEAD_DIM;
        #pragma unroll
        for (int c = 0; c < 2; ++c)
            qf[c] = ld16(qp + c * 32 + half16 * 8, qp + c * 32 + 16 + half16 * 8);
    }

    // Prime the pipeline: TDM loads for chunk 0 into buffer 0 (wave 0 owns TDM).
    if (wave == 0) {
        // K chunk: 32 rows x 64 halves, row stride 64; pad 4 DW every 32 DW -> 72
        tdm_load_2d_f16(Kh, (unsigned)(size_t)&Kbuf[0][0],
                        64, SK, 64, 32, 64, /*interval 32DW*/4, /*amount 4DW*/3);
        // V^T chunk: 64 rows x 32 halves, row stride SK; pad 4 DW every 16 DW -> 40
        tdm_load_2d_f16(Vh, (unsigned)(size_t)&Vbuf[0][0],
                        SK, 64, 32, 64, SK, /*interval 16DW*/3, /*amount 4DW*/3);
    }

    float mrow[8], lrow[8];
    v8f acc[4];
    #pragma unroll
    for (int v = 0; v < 8; ++v) { mrow[v] = -3.0e38f; lrow[v] = 0.0f; }
    #pragma unroll
    for (int i = 0; i < 4; ++i) acc[i] = zero8();

    for (int k0 = 0; k0 < SK; k0 += 32) {
        const int buf = (k0 >> 5) & 1;
        if (wave == 0) {
            if (k0 + 32 < SK) {
                // issue next chunk into the other buffer, then retire current pair:
                // TDM completes in order, so waiting to <=2 outstanding means the
                // current chunk's K+V pair has landed in LDS.
                tdm_load_2d_f16(Kh + (size_t)(k0 + 32) * HEAD_DIM,
                                (unsigned)(size_t)&Kbuf[buf ^ 1][0],
                                64, SK, 64, 32, 64, 4, 3);
                tdm_load_2d_f16(Vh + (k0 + 32),
                                (unsigned)(size_t)&Vbuf[buf ^ 1][0],
                                SK, 64, 32, 64, SK, 3, 3);
                __builtin_amdgcn_s_wait_tensorcnt(2);
            } else {
                __builtin_amdgcn_s_wait_tensorcnt(0);
            }
        }
        __syncthreads();   // publish TDM-written LDS to all waves

        const half_t* Kc = &Kbuf[buf][0];
        const half_t* Vc = &Vbuf[buf][0];

        // S = Q K^T : load all 4 K-fragments, then 4 WMMAs (2 indep chains)
        v16h kfr[2][2];
        #pragma unroll
        for (int ns = 0; ns < 2; ++ns) {
            int kcol = ns * 16 + l16;
            #pragma unroll
            for (int dc = 0; dc < 2; ++dc)
                kfr[ns][dc] = ld16(&Kc[kcol * 72 + dc * 32 + half16 * 8],
                                   &Kc[kcol * 72 + dc * 32 + 16 + half16 * 8]);
        }
        v8f s[2];
        #pragma unroll
        for (int ns = 0; ns < 2; ++ns) {
            v8f sa = zero8();
            sa = __builtin_amdgcn_wmma_f32_16x16x32_f16(
                false, qf[0], false, kfr[ns][0], (short)0, sa, false, false);
            sa = __builtin_amdgcn_wmma_f32_16x16x32_f16(
                false, qf[1], false, kfr[ns][1], (short)0, sa, false, false);
            s[ns] = sa;
        }

        // Online softmax. C-layout: row m = v + half16*8 in VGPR slot v,
        // columns across the 16 lanes of each half -> xor-reduce masks 1,2,4,8.
        const float sc = 0.125f;   // 1/sqrt(64)
        float mnew[8], scale[8];
        #pragma unroll
        for (int v = 0; v < 8; ++v) {
            float s0 = s[0][v] * sc, s1 = s[1][v] * sc;
            s[0][v] = s0; s[1][v] = s1;
            float mx = fmaxf(s0, s1);
            mx = fmaxf(mx, __shfl_xor(mx, 1, 32));
            mx = fmaxf(mx, __shfl_xor(mx, 2, 32));
            mx = fmaxf(mx, __shfl_xor(mx, 4, 32));
            mx = fmaxf(mx, __shfl_xor(mx, 8, 32));
            mnew[v]  = fmaxf(mrow[v], mx);
            scale[v] = __expf(mrow[v] - mnew[v]);
        }

        half_t* Pw = &Pbuf[wave * 16 * 40];
        #pragma unroll
        for (int v = 0; v < 8; ++v) {
            float p0 = __expf(s[0][v] - mnew[v]);
            float p1 = __expf(s[1][v] - mnew[v]);
            float rs = p0 + p1;
            rs += __shfl_xor(rs, 1, 32);
            rs += __shfl_xor(rs, 2, 32);
            rs += __shfl_xor(rs, 4, 32);
            rs += __shfl_xor(rs, 8, 32);
            lrow[v] = lrow[v] * scale[v] + rs;
            mrow[v] = mnew[v];
            int m = v + half16 * 8;
            Pw[m * 40 + l16]      = (half_t)p0;     // C-layout -> [m][k] in LDS
            Pw[m * 40 + 16 + l16] = (half_t)p1;
            #pragma unroll
            for (int t = 0; t < 4; ++t) acc[t][v] *= scale[v];
        }

        // Re-read P as A-fragment (wave-private region, in-order LDS per wave)
        v16h pf = ld16(&Pw[l16 * 40 + half16 * 8], &Pw[l16 * 40 + 16 + half16 * 8]);

        // O += P V : load all 4 V-fragments, then 4 independent WMMAs
        v16h vfr[4];
        #pragma unroll
        for (int nd = 0; nd < 4; ++nd) {
            int drow = nd * 16 + l16;
            vfr[nd] = ld16(&Vc[drow * 40 + half16 * 8],
                           &Vc[drow * 40 + 16 + half16 * 8]);
        }
        #pragma unroll
        for (int nd = 0; nd < 4; ++nd)
            acc[nd] = __builtin_amdgcn_wmma_f32_16x16x32_f16(
                false, pf, false, vfr[nd], (short)0, acc[nd], false, false);
        __syncthreads();   // all waves done with buf before TDM overwrites it
    }

    // Normalize and store ctx f16, head-major concat: col = h*64 + d
    half_t* C = Ctx + ((size_t)b * SQ + qtile * 64 + wave * 16) * EMBED + h * HEAD_DIM;
    #pragma unroll
    for (int nd = 0; nd < 4; ++nd) {
        int n = nd * 16 + l16;
        #pragma unroll
        for (int v = 0; v < 8; ++v) {
            int m = v + half16 * 8;
            C[m * EMBED + n] = (half_t)(acc[nd][v] / lrow[v]);
        }
    }
}

// ---------------------------------------------------------------------------
// Kernel 3: output projection  Out = Ctx (4096x1024 f16) * Wo + bo  (f32 out)
// ---------------------------------------------------------------------------
__global__ __launch_bounds__(128)
void outproj_kernel(const half_t* __restrict__ Ctx, const float* __restrict__ Wo,
                    const float* __restrict__ bo, float* __restrict__ Out)
{
    __shared__ half_t As[64 * 40];
    __shared__ half_t Bt[64 * 40];

    const int tid = threadIdx.x;
    const int wave = tid >> 5, lane = tid & 31;
    const int half16 = lane >> 4, l16 = lane & 15;
    const int m0 = blockIdx.x * 64;   // 64 m-tiles over B*SQ=4096
    const int n0 = blockIdx.y * 64;   // 16 n-tiles over EMBED

    v8f acc[4];
    #pragma unroll
    for (int i = 0; i < 4; ++i) acc[i] = zero8();

    for (int k0 = 0; k0 < EMBED; k0 += 32) {
        {   // A: f16 straight copy (b128 in / b128 out)
            int r = tid >> 1, cc = (tid & 1) * 16;
            const half_t* src = Ctx + (size_t)(m0 + r) * EMBED + k0 + cc;
            *(v8h*)&As[r * 40 + cc]     = *(const v8h*)src;
            *(v8h*)&As[r * 40 + cc + 8] = *(const v8h*)(src + 8);
            if (k0 + 32 < EMBED) __builtin_prefetch(src + 32, 0, 1);
        }
        {   // B^T: thread owns column n of Wo; coalesced strided reads,
            // packed contiguous ds_store_b128 x2 writes
            int n = tid & 63, rh = tid >> 6;
            const float* src = Wo + (size_t)(k0 + rh * 16) * EMBED + n0 + n;
            v8h h0, h1;
            #pragma unroll
            for (int j = 0; j < 8; ++j) h0[j] = (half_t)src[(size_t)j * EMBED];
            #pragma unroll
            for (int j = 0; j < 8; ++j) h1[j] = (half_t)src[(size_t)(8 + j) * EMBED];
            *(v8h*)&Bt[n * 40 + rh * 16]     = h0;
            *(v8h*)&Bt[n * 40 + rh * 16 + 8] = h1;
            if (k0 + 32 < EMBED) __builtin_prefetch(src + (size_t)32 * EMBED, 0, 1);
        }
        __syncthreads();

        int mrow = wave * 16 + l16;
        v16h af = ld16(&As[mrow * 40 + half16 * 8], &As[mrow * 40 + 16 + half16 * 8]);
        v16h bf[4];
        #pragma unroll
        for (int ns = 0; ns < 4; ++ns) {
            int ncol = ns * 16 + l16;
            bf[ns] = ld16(&Bt[ncol * 40 + half16 * 8], &Bt[ncol * 40 + 16 + half16 * 8]);
        }
        #pragma unroll
        for (int ns = 0; ns < 4; ++ns)
            acc[ns] = __builtin_amdgcn_wmma_f32_16x16x32_f16(
                false, af, false, bf[ns], (short)0, acc[ns], false, false);
        __syncthreads();
    }

    #pragma unroll
    for (int ns = 0; ns < 4; ++ns) {
        int n = n0 + ns * 16 + l16;
        float bb = bo[n];
        #pragma unroll
        for (int v = 0; v < 8; ++v) {
            int m = m0 + wave * 16 + v + half16 * 8;
            Out[(size_t)m * EMBED + n] = acc[ns][v] + bb;
        }
    }
}

// ---------------------------------------------------------------------------
extern "C" void kernel_launch(void* const* d_in, const int* in_sizes, int n_in,
                              void* d_out, int out_size, void* d_ws, size_t ws_size,
                              hipStream_t stream)
{
    (void)in_sizes; (void)n_in; (void)out_size; (void)ws_size;
    const float* dec = (const float*)d_in[0];
    const float* enc = (const float*)d_in[1];
    const float* wq  = (const float*)d_in[2];
    const float* bq  = (const float*)d_in[3];
    const float* wk  = (const float*)d_in[4];
    const float* bk  = (const float*)d_in[5];
    const float* wv  = (const float*)d_in[6];
    const float* bv  = (const float*)d_in[7];
    const float* wo  = (const float*)d_in[8];
    const float* bo  = (const float*)d_in[9];
    float* out = (float*)d_out;

    const size_t NQ = (size_t)BATCH * HEADS * SQ * HEAD_DIM;   // 4,194,304
    half_t* Q   = (half_t*)d_ws;
    half_t* K   = Q + NQ;
    half_t* Vt  = K + NQ;
    half_t* Ctx = Vt + NQ;   // [B][SQ][EMBED] = same count

    dim3 blk(128);
    proj_kernel<<<dim3(SQ / 64, BATCH * HEADS), blk, 0, stream>>>(dec, wq, bq, Q,  SQ, 0);
    proj_kernel<<<dim3(SK / 64, BATCH * HEADS), blk, 0, stream>>>(enc, wk, bk, K,  SK, 0);
    proj_kernel<<<dim3(SK / 64, BATCH * HEADS), blk, 0, stream>>>(enc, wv, bv, Vt, SK, 1);
    attn_kernel<<<dim3(SQ / 64, BATCH * HEADS), blk, 0, stream>>>(Q, K, Vt, Ctx);
    outproj_kernel<<<dim3(BATCH * SQ / 64, EMBED / 64), blk, 0, stream>>>(Ctx, wo, bo, out);
}
